// NNUENet_6846177869883
// MI455X (gfx1250) — matrix operand order repaired and checked by reading, not compiled, not run on hardware
//
#include <hip/hip_runtime.h>

typedef _Float16 v16h __attribute__((ext_vector_type(16)));
typedef _Float16 v8h  __attribute__((ext_vector_type(8)));
typedef float    v8f  __attribute__((ext_vector_type(8)));
typedef float    v4f  __attribute__((ext_vector_type(4)));

#define FEAT  1260
#define NACC  128
#define BROWS 64          // batch rows per block
#define KC    32          // WMMA K chunk
#define NKC   40          // ceil(1260/32), tail zero-padded
#define SA    40          // LDS stride (halfs) for 32-wide K tiles (80B: 16B aligned, 20 banks)
#define SX    264         // LDS stride (halfs) for 256-wide x / fc1 weights (528B: conflict-free)

// LDS layout (in halfs). region0 is phase-aliased:
//   GEMM phase: A_stm[64*40]@0, A_nstm[64*40]@2560, B[128*40]@5120   (uses 0..10239)
//   tail phase: fc1w[32*264]@0, h1[64*40]@8448, fc2w[32*40]@11008, h2[64*40]@12288 (0..14847)
// x[64*264] lives at 14848 (written at end of GEMM phase, disjoint from region0).
#define OFF_ASTM   0
#define OFF_ANSTM  2560
#define OFF_B      5120
#define OFF_FC1W   0
#define OFF_H1     8448
#define OFF_FC2W   11008
#define OFF_H2     12288
#define OFF_X      14848
#define LDS_HALFS  31744   // 63,488 bytes

__device__ __forceinline__ v16h ldfrag(const _Float16* p0, const _Float16* p1) {
    v8h a = *(const v8h*)p0;
    v8h b = *(const v8h*)p1;
    return __builtin_shufflevector(a, b, 0,1,2,3,4,5,6,7,8,9,10,11,12,13,14,15);
}

#define WMMA_F16(A, B, C) \
    __builtin_amdgcn_wmma_f32_16x16x32_f16(false, (A), false, (B), (short)0, (C), false, false)

__device__ __forceinline__ float crelu(float x) {
    return fminf(fmaxf(x, 0.0f), 1.0f);
}

__global__ __launch_bounds__(256)
void nnue_fused_kernel(const float* __restrict__ stm, const float* __restrict__ nstm,
                       const float* __restrict__ accw, const float* __restrict__ accb,
                       const float* __restrict__ fc1w, const float* __restrict__ fc1b,
                       const float* __restrict__ fc2w, const float* __restrict__ fc2b,
                       const float* __restrict__ outw, const float* __restrict__ outb,
                       float* __restrict__ out)
{
    __shared__ _Float16 sm[LDS_HALFS];

    const int tid  = threadIdx.x;
    const int wave = tid >> 5;
    const int lane = tid & 31;
    const int l15  = lane & 15;
    const int lhi  = lane >> 4;
    const long rowbase = (long)blockIdx.x * BROWS;

    // ---------------- accumulator layer: [64 x 1260] x [1260 x 128], 2 perspectives ----
    const int wm = wave & 1;   // 0..1 : 32-row slab
    const int wn = wave >> 1;  // 0..3 : 32-col slab of the 128 outputs

    v8f acc[2][2][2];          // [persp][mi][ni]
    #pragma unroll
    for (int p = 0; p < 2; ++p)
        #pragma unroll
        for (int mi = 0; mi < 2; ++mi)
            #pragma unroll
            for (int ni = 0; ni < 2; ++ni)
                acc[p][mi][ni] = v8f{};

    // staging assignments: A tiles 64x32 (8 floats/thread), B tile 128x32 (16 floats/thread)
    const int ar = tid >> 2;          // 0..63
    const int ak = (tid & 3) * 8;     // 0,8,16,24
    const int bn = tid >> 1;          // 0..127
    const int bk = (tid & 1) * 16;    // 0,16

    float ra_s[8], ra_n[8], rb[16];

    auto loadA = [&](const float* __restrict__ f, float* r, int kb) {
        const float* p = f + (rowbase + ar) * FEAT + kb + ak;
        if (kb + KC <= FEAT) {
            v4f x0 = *(const v4f*)(p);
            v4f x1 = *(const v4f*)(p + 4);
            #pragma unroll
            for (int j = 0; j < 4; ++j) { r[j] = x0[j]; r[4 + j] = x1[j]; }
        } else {
            #pragma unroll
            for (int j = 0; j < 8; ++j) {
                int k = kb + ak + j;
                r[j] = (k < FEAT) ? p[j] : 0.0f;
            }
        }
    };
    auto loadB = [&](int kb) {
        const float* p = accw + (long)bn * FEAT + kb + bk;
        if (kb + KC <= FEAT) {
            #pragma unroll
            for (int q = 0; q < 4; ++q) {
                v4f x = *(const v4f*)(p + q * 4);
                #pragma unroll
                for (int j = 0; j < 4; ++j) rb[q * 4 + j] = x[j];
            }
        } else {
            #pragma unroll
            for (int j = 0; j < 16; ++j) {
                int k = kb + bk + j;
                rb[j] = (k < FEAT) ? p[j] : 0.0f;
            }
        }
    };

    loadA(stm,  ra_s, 0);
    loadA(nstm, ra_n, 0);
    loadB(0);

    for (int kc = 0; kc < NKC; ++kc) {
        // commit staged chunk to LDS (f32 -> f16 conversion happens here)
        {
            _Float16* da = sm + OFF_ASTM  + ar * SA + ak;
            _Float16* dn = sm + OFF_ANSTM + ar * SA + ak;
            #pragma unroll
            for (int j = 0; j < 8; ++j) { da[j] = (_Float16)ra_s[j]; dn[j] = (_Float16)ra_n[j]; }
            _Float16* db = sm + OFF_B + bn * SA + bk;
            #pragma unroll
            for (int j = 0; j < 16; ++j) db[j] = (_Float16)rb[j];
        }
        __syncthreads();

        // issue next chunk's global loads so VMEM overlaps the WMMAs below
        if (kc + 1 < NKC) {
            int kb = (kc + 1) * KC;
            loadA(stm,  ra_s, kb);
            loadA(nstm, ra_n, kb);
            loadB(kb);
        }

        // fragments (ISA §7.12.2 layouts)
        v16h bf[2], as[2], an[2];
        #pragma unroll
        for (int ni = 0; ni < 2; ++ni) {
            const _Float16* p0 = sm + OFF_B + (wn * 32 + ni * 16 + l15) * SA + lhi * 8;
            bf[ni] = ldfrag(p0, p0 + 16);
        }
        #pragma unroll
        for (int mi = 0; mi < 2; ++mi) {
            const _Float16* ps = sm + OFF_ASTM  + (wm * 32 + mi * 16 + l15) * SA + lhi * 8;
            const _Float16* pn = sm + OFF_ANSTM + (wm * 32 + mi * 16 + l15) * SA + lhi * 8;
            as[mi] = ldfrag(ps, ps + 16);
            an[mi] = ldfrag(pn, pn + 16);
        }
        #pragma unroll
        for (int mi = 0; mi < 2; ++mi)
            #pragma unroll
            for (int ni = 0; ni < 2; ++ni) {
                acc[0][mi][ni] = WMMA_F16(as[mi], bf[ni], acc[0][mi][ni]);
                acc[1][mi][ni] = WMMA_F16(an[mi], bf[ni], acc[1][mi][ni]);
            }
        __syncthreads();
    }

    // bias + crelu, write concat [stm||nstm] as f16 x[64][256] (stride SX)
    #pragma unroll
    for (int p = 0; p < 2; ++p)
        #pragma unroll
        for (int mi = 0; mi < 2; ++mi)
            #pragma unroll
            for (int ni = 0; ni < 2; ++ni) {
                int n = wn * 32 + ni * 16 + l15;
                float b = accb[n];
                int col = p * NACC + n;
                int r0 = wm * 32 + mi * 16 + lhi * 8;
                #pragma unroll
                for (int v = 0; v < 8; ++v)
                    sm[OFF_X + (r0 + v) * SX + col] = (_Float16)crelu(acc[p][mi][ni][v] + b);
            }

    // stage fc1/fc2 weights into (now free) region0
    {
        int n = tid >> 3, k0 = (tid & 7) * 32;       // fc1w: 32x256
        const float* p = fc1w + n * 256 + k0;
        #pragma unroll
        for (int q = 0; q < 8; ++q) {
            v4f x = *(const v4f*)(p + q * 4);
            #pragma unroll
            for (int j = 0; j < 4; ++j)
                sm[OFF_FC1W + n * SX + k0 + q * 4 + j] = (_Float16)x[j];
        }
        int k2 = (tid & 7) * 4;                       // fc2w: 32x32
        v4f y = *(const v4f*)(fc2w + n * 32 + k2);
        #pragma unroll
        for (int j = 0; j < 4; ++j)
            sm[OFF_FC2W + n * SA + k2 + j] = (_Float16)y[j];
    }
    __syncthreads();

    // ---------------- fc1: [64 x 256] x [256 x 32] ----------------
    const int wm2 = wave & 3;   // 0..3 : 16-row tile
    const int wn2 = wave >> 2;  // 0..1 : 16-col tile
    {
        v8f h = v8f{};
        #pragma unroll
        for (int kc = 0; kc < 8; ++kc) {
            const _Float16* pa = sm + OFF_X    + (wm2 * 16 + l15) * SX + kc * 32 + lhi * 8;
            const _Float16* pb = sm + OFF_FC1W + (wn2 * 16 + l15) * SX + kc * 32 + lhi * 8;
            h = WMMA_F16(ldfrag(pa, pa + 16), ldfrag(pb, pb + 16), h);
        }
        int n = wn2 * 16 + l15;
        float b = fc1b[n];
        int r0 = wm2 * 16 + lhi * 8;
        #pragma unroll
        for (int v = 0; v < 8; ++v)
            sm[OFF_H1 + (r0 + v) * SA + n] = (_Float16)crelu(h[v] + b);
    }
    __syncthreads();

    // ---------------- fc2: [64 x 32] x [32 x 32] ----------------
    {
        const _Float16* pa = sm + OFF_H1   + (wm2 * 16 + l15) * SA + lhi * 8;
        const _Float16* pb = sm + OFF_FC2W + (wn2 * 16 + l15) * SA + lhi * 8;
        v8f g = v8f{};
        g = WMMA_F16(ldfrag(pa, pa + 16), ldfrag(pb, pb + 16), g);
        int n = wn2 * 16 + l15;
        float b = fc2b[n];
        int r0 = wm2 * 16 + lhi * 8;
        #pragma unroll
        for (int v = 0; v < 8; ++v)
            sm[OFF_H2 + (r0 + v) * SA + n] = (_Float16)crelu(g[v] + b);
    }
    __syncthreads();

    // ---------------- output: dot(32) + sigmoid ----------------
    if (tid < BROWS) {
        float s = outb[0];
        #pragma unroll
        for (int k = 0; k < 32; ++k)
            s += outw[k] * (float)sm[OFF_H2 + tid * SA + k];
        out[rowbase + tid] = 1.0f / (1.0f + __expf(-s));
    }
}

extern "C" void kernel_launch(void* const* d_in, const int* in_sizes, int n_in,
                              void* d_out, int out_size, void* d_ws, size_t ws_size,
                              hipStream_t stream) {
    (void)in_sizes; (void)n_in; (void)d_ws; (void)ws_size;
    const float* stm  = (const float*)d_in[0];
    const float* nstm = (const float*)d_in[1];
    const float* accw = (const float*)d_in[2];
    const float* accb = (const float*)d_in[3];
    const float* f1w  = (const float*)d_in[4];
    const float* f1b  = (const float*)d_in[5];
    const float* f2w  = (const float*)d_in[6];
    const float* f2b  = (const float*)d_in[7];
    const float* ow   = (const float*)d_in[8];
    const float* ob   = (const float*)d_in[9];
    float* out = (float*)d_out;

    dim3 grid(out_size / BROWS);   // 65536 / 64 = 1024 blocks
    dim3 block(256);               // 8 wave32 waves
    nnue_fused_kernel<<<grid, block, 0, stream>>>(stm, nstm, accw, accb,
                                                  f1w, f1b, f2w, f2b, ow, ob, out);
}
